// CompositeGraphNetWithFC_8151847928475
// MI455X (gfx1250) — matrix operand order repaired and checked by reading, not compiled, not run on hardware
//
#include <hip/hip_runtime.h>

typedef __attribute__((ext_vector_type(16))) _Float16 v16h;
typedef __attribute__((ext_vector_type(8)))  float    v8f;

#define HID 128

// ---------------- elementwise / utility kernels ----------------

__global__ void k_fill(float* __restrict__ p, float v, int n) {
  int i = blockIdx.x * 256 + threadIdx.x;
  if (i < n) p[i] = v;
}

__global__ void k_deg_scatter(const int* __restrict__ col, const float* __restrict__ ew,
                              float* __restrict__ deg, int E) {
  int e = blockIdx.x * 256 + threadIdx.x;
  if (e < E) unsafeAtomicAdd(&deg[col[e]], ew[e]);
}

__global__ void k_rsqrt(const float* __restrict__ deg, float* __restrict__ dis, int n) {
  int i = blockIdx.x * 256 + threadIdx.x;
  if (i < n) { float d = deg[i]; dis[i] = d > 0.f ? rsqrtf(d) : 0.f; }
}

// counts D (per node), Bc (per hyperedge) and nhw accumulation in one pass
__global__ void k_hyper_count(const int* __restrict__ nodei, const int* __restrict__ hei,
                              const float* __restrict__ hd,
                              float* __restrict__ D, float* __restrict__ Bc,
                              float* __restrict__ nhw, int I) {
  int i = blockIdx.x * 256 + threadIdx.x;
  if (i < I) {
    int v = nodei[i], h = hei[i];
    unsafeAtomicAdd(&D[v], 1.0f);
    unsafeAtomicAdd(&Bc[h], 1.0f);
    unsafeAtomicAdd(&nhw[v], hd[h]);
  }
}

__global__ void k_invert(float* __restrict__ p, int n) {
  int i = blockIdx.x * 256 + threadIdx.x;
  if (i < n) { float d = p[i]; p[i] = d > 0.f ? 1.0f / d : 0.f; }
}

// W (K x Ncol, row-major fp32)  ->  Wt (Ncol x K, f16)  [N-major so B-frag loads are contiguous]
__global__ void k_wconv(const float* __restrict__ W, _Float16* __restrict__ Wt, int K, int Ncol) {
  int i = blockIdx.x * 256 + threadIdx.x;
  if (i < K * Ncol) {
    int n = i / K, k = i - n * K;
    Wt[i] = (_Float16)W[(size_t)k * Ncol + n];
  }
}

// out[i] = dis[row]^2 * h[i]   (GCN self-loop term: norm = dis*1*dis)
__global__ void k_selfloop_init(const float* __restrict__ h, const float* __restrict__ dis,
                                float* __restrict__ out, int n) {
  int i = blockIdx.x * 256 + threadIdx.x;
  if (i < n) { float d = dis[i >> 7]; out[i] = d * d * h[i]; }
}

__global__ void k_bias_relu(float* __restrict__ p, const float* __restrict__ b, int n) {
  int i = blockIdx.x * 256 + threadIdx.x;
  if (i < n) p[i] = fmaxf(p[i] + b[i & (HID - 1)], 0.f);
}

__global__ void k_scale_rows(const float* __restrict__ x, const float* __restrict__ s,
                             float* __restrict__ out, int n) {
  int i = blockIdx.x * 256 + threadIdx.x;
  if (i < n) out[i] = x[i] * s[i >> 7];
}

// out[i] = relu(Dinv[row]*out[i] + b[col])
__global__ void k_hyper_fin(float* __restrict__ p, const float* __restrict__ Dinv,
                            const float* __restrict__ b, int n) {
  int i = blockIdx.x * 256 + threadIdx.x;
  if (i < n) p[i] = fmaxf(Dinv[i >> 7] * p[i] + b[i & (HID - 1)], 0.f);
}

// ---------------- scatter kernels: one wave32 per edge/incidence ----------------

__global__ __launch_bounds__(256)
void k_gcn_scatter(const float* __restrict__ h, const int* __restrict__ row,
                   const int* __restrict__ col, const float* __restrict__ ew,
                   const float* __restrict__ dis, float* __restrict__ out, int E) {
  int e = blockIdx.x * 8 + (threadIdx.x >> 5);
  if (e >= E) return;
  int lane = threadIdx.x & 31;
  int r = row[e], c = col[e];
  float coeff = dis[r] * ew[e] * dis[c];
  const float4* hr = (const float4*)(h + (size_t)r * HID);
  float4 v = hr[lane];                      // 32 lanes x float4 = 128 features
  float* o = out + (size_t)c * HID + lane * 4;
  unsafeAtomicAdd(o + 0, coeff * v.x);
  unsafeAtomicAdd(o + 1, coeff * v.y);
  unsafeAtomicAdd(o + 2, coeff * v.z);
  unsafeAtomicAdd(o + 3, coeff * v.w);
}

__global__ __launch_bounds__(256)
void k_scatter_n2e(const float* __restrict__ h, const int* __restrict__ nodei,
                   const int* __restrict__ hei, float* __restrict__ ef, int I) {
  int i = blockIdx.x * 8 + (threadIdx.x >> 5);
  if (i >= I) return;
  int lane = threadIdx.x & 31;
  int v = nodei[i], he = hei[i];
  const float4* hv = (const float4*)(h + (size_t)v * HID);
  float4 x = hv[lane];
  float* o = ef + (size_t)he * HID + lane * 4;
  unsafeAtomicAdd(o + 0, x.x);
  unsafeAtomicAdd(o + 1, x.y);
  unsafeAtomicAdd(o + 2, x.z);
  unsafeAtomicAdd(o + 3, x.w);
}

__global__ __launch_bounds__(256)
void k_scatter_e2n(const float* __restrict__ ef, const float* __restrict__ Binv,
                   const int* __restrict__ nodei, const int* __restrict__ hei,
                   float* __restrict__ out, int I) {
  int i = blockIdx.x * 8 + (threadIdx.x >> 5);
  if (i >= I) return;
  int lane = threadIdx.x & 31;
  int v = nodei[i], he = hei[i];
  float cb = Binv[he];
  const float4* hv = (const float4*)(ef + (size_t)he * HID);
  float4 x = hv[lane];
  float* o = out + (size_t)v * HID + lane * 4;
  unsafeAtomicAdd(o + 0, cb * x.x);
  unsafeAtomicAdd(o + 1, cb * x.y);
  unsafeAtomicAdd(o + 2, cb * x.z);
  unsafeAtomicAdd(o + 3, cb * x.w);
}

// ---------------- WMMA GEMM: OUT[N x 128] = act( [A0|A1] @ Wt^T + bias ) ----------------
// 256 threads = 8 waves; block covers 16 rows; wave w covers output cols [16w,16w+16).
// Wt is (128 x K) f16, N-major. K = K0 + K1, multiple of 32, <= 256.

__global__ __launch_bounds__(256)
void k_gemm_wmma(const float* __restrict__ A0, const float* __restrict__ A1,
                 int K0, int K1,
                 const _Float16* __restrict__ Wt, const float* __restrict__ bias,
                 float* __restrict__ OUT, int relu, int Nrows) {
  const int K = K0 + K1;
  const int stride = K + 8;                  // pad to dodge LDS bank conflicts
  __shared__ _Float16 Atile[16 * (256 + 8)];
  int rowBase = blockIdx.x * 16;

  for (int idx = threadIdx.x; idx < 16 * K; idx += 256) {
    int r = idx / K, k = idx - r * K;
    int gr = rowBase + r;
    float v = 0.f;
    if (gr < Nrows) v = (k < K0) ? A0[(size_t)gr * K0 + k] : A1[(size_t)gr * K1 + (k - K0)];
    Atile[r * stride + k] = (_Float16)v;
  }
  __syncthreads();

  int wave = threadIdx.x >> 5;
  int lane = threadIdx.x & 31;
  int half = lane >> 4;                      // 0: lanes 0-15, 1: lanes 16-31
  int nn   = lane & 15;                      // M for A-frag, N for B/C-frag

  v8f acc = {0.f, 0.f, 0.f, 0.f, 0.f, 0.f, 0.f, 0.f};
  const _Float16* wcol = Wt + (size_t)(wave * 16 + nn) * K;
  const _Float16* arow = Atile + nn * stride;

  for (int kb = 0; kb < K; kb += 32) {
    v16h a, b;
#pragma unroll
    for (int v = 0; v < 8; ++v) {
      // A 16x32 f16 layout: VGPR v<4 -> K = 8*half + 2v{,+1}; v>=4 -> K = 16 + 8*half + 2(v-4)
      int ka = kb + ((v < 4) ? (half * 8 + 2 * v) : (16 + half * 8 + 2 * (v - 4)));
      a[2 * v]     = arow[ka];
      a[2 * v + 1] = arow[ka + 1];
      // B 32x16 f16 layout: same K order as A: v<4 -> K = 8*half + 2v{,+1}; v>=4 -> K = 16 + 8*half + 2(v-4)
      int kbn = kb + ((v < 4) ? (half * 8 + 2 * v) : (16 + half * 8 + 2 * (v - 4)));
      b[2 * v]     = wcol[kbn];
      b[2 * v + 1] = wcol[kbn + 1];
    }
    acc = __builtin_amdgcn_wmma_f32_16x16x32_f16(false, a, false, b, (short)0, acc,
                                                 false, false);
  }

  float bv = bias ? bias[wave * 16 + nn] : 0.f;
#pragma unroll
  for (int r = 0; r < 8; ++r) {
    int gr = rowBase + r + half * 8;         // C/D layout: M = r + 8*(lane>=16)
    if (gr < Nrows) {
      float v = acc[r] + bv;
      if (relu) v = fmaxf(v, 0.f);
      OUT[(size_t)gr * HID + wave * 16 + nn] = v;
    }
  }
}

// ---------------- final 128 -> 2 projection ----------------

__global__ void k_outproj(const float* __restrict__ h, const float* __restrict__ W,
                          const float* __restrict__ b, float* __restrict__ out, int Nn) {
  int n = blockIdx.x * 256 + threadIdx.x;
  if (n >= Nn) return;
  const float* hn = h + (size_t)n * HID;
  float a0 = b[0], a1 = b[1];
#pragma unroll 4
  for (int k = 0; k < HID; ++k) {
    float x = hn[k];
    a0 += x * W[2 * k];
    a1 += x * W[2 * k + 1];
  }
  out[2 * n]     = a0;
  out[2 * n + 1] = a1;
}

// ---------------- launcher ----------------

extern "C" void kernel_launch(void* const* d_in, const int* in_sizes, int n_in,
                              void* d_out, int out_size, void* d_ws, size_t ws_size,
                              hipStream_t stream) {
  (void)n_in; (void)out_size; (void)ws_size;
  const float* x   = (const float*)d_in[0];
  const int*   ei  = (const int*)d_in[1];
  const float* ew  = (const float*)d_in[2];
  const int*   hi  = (const int*)d_in[3];
  const float* hd  = (const float*)d_in[4];
  const float* Wg1 = (const float*)d_in[5];  const float* bg1 = (const float*)d_in[6];
  const float* Wg2 = (const float*)d_in[7];  const float* bg2 = (const float*)d_in[8];
  const float* Wh1 = (const float*)d_in[9];  const float* bh1 = (const float*)d_in[10];
  const float* Wh2 = (const float*)d_in[11]; const float* bh2 = (const float*)d_in[12];
  const float* Wfc = (const float*)d_in[13]; const float* bfc = (const float*)d_in[14];
  const float* Wo  = (const float*)d_in[15]; const float* bo  = (const float*)d_in[16];

  const int N = in_sizes[0] / HID;
  const int E = in_sizes[1] / 2;
  const int I = in_sizes[3] / 2;
  const int M = in_sizes[4];

  const int* row   = ei;        // edge_index[0]
  const int* colv  = ei + E;    // edge_index[1]
  const int* nodei = hi;        // hyperedge_index[0]
  const int* hei   = hi + I;    // hyperedge_index[1]

  size_t NB = (size_t)N * HID;
  float* B1   = (float*)d_ws;
  float* B2   = B1 + NB;
  float* B3   = B2 + NB;
  float* ef   = B3 + NB;                 // M x 128
  float* deg  = ef + (size_t)M * HID;    // N
  float* dis  = deg + N;                 // N
  float* Dinv = dis + N;                 // N (counts, then inverted)
  float* nhw  = Dinv + N;                // N
  float* Binv = nhw + N;                 // M (counts, then inverted)
  _Float16* Wg1t = (_Float16*)(Binv + M);
  _Float16* Wg2t = Wg1t + 128 * 128;
  _Float16* Wh1t = Wg2t + 128 * 128;
  _Float16* Wh2t = Wh1t + 128 * 128;
  _Float16* Wfct = Wh2t + 128 * 128;     // 128 x 256

  dim3 blk(256);
  auto g1 = [](int n) { return dim3((unsigned)((n + 255) / 256)); };
  const dim3 gGemm((unsigned)((N + 15) / 16));
  const dim3 gE((unsigned)((E + 7) / 8));
  const dim3 gI((unsigned)((I + 7) / 8));
  const int  nNB = (int)NB;
  const int  nMB = M * HID;

  // --- normalization precompute ---
  k_fill<<<g1(N), blk, 0, stream>>>(deg, 1.0f, N);        // self-loop weight 1
  k_fill<<<g1(N), blk, 0, stream>>>(Dinv, 0.f, N);
  k_fill<<<g1(M), blk, 0, stream>>>(Binv, 0.f, M);
  k_fill<<<g1(N), blk, 0, stream>>>(nhw, 0.f, N);
  k_deg_scatter<<<g1(E), blk, 0, stream>>>(colv, ew, deg, E);
  k_rsqrt<<<g1(N), blk, 0, stream>>>(deg, dis, N);
  k_hyper_count<<<g1(I), blk, 0, stream>>>(nodei, hei, hd, Dinv, Binv, nhw, I);
  k_invert<<<g1(N), blk, 0, stream>>>(Dinv, N);
  k_invert<<<g1(M), blk, 0, stream>>>(Binv, M);

  // --- weight conversion (fp32 KxN -> f16 NxK) ---
  k_wconv<<<g1(128 * 128), blk, 0, stream>>>(Wg1, Wg1t, 128, 128);
  k_wconv<<<g1(128 * 128), blk, 0, stream>>>(Wg2, Wg2t, 128, 128);
  k_wconv<<<g1(128 * 128), blk, 0, stream>>>(Wh1, Wh1t, 128, 128);
  k_wconv<<<g1(128 * 128), blk, 0, stream>>>(Wh2, Wh2t, 128, 128);
  k_wconv<<<g1(256 * 128), blk, 0, stream>>>(Wfc, Wfct, 256, 128);

  // --- GCN layer 1: B2 = relu(agg(x @ Wg1) + bg1) ---
  k_gemm_wmma<<<gGemm, blk, 0, stream>>>(x, nullptr, 128, 0, Wg1t, nullptr, B1, 0, N);
  k_selfloop_init<<<g1(nNB), blk, 0, stream>>>(B1, dis, B2, nNB);
  k_gcn_scatter<<<gE, blk, 0, stream>>>(B1, row, colv, ew, dis, B2, E);
  k_bias_relu<<<g1(nNB), blk, 0, stream>>>(B2, bg1, nNB);
  // --- GCN layer 2: B1 = relu(agg(B2 @ Wg2) + bg2)  -> x1 ---
  k_gemm_wmma<<<gGemm, blk, 0, stream>>>(B2, nullptr, 128, 0, Wg2t, nullptr, B3, 0, N);
  k_selfloop_init<<<g1(nNB), blk, 0, stream>>>(B3, dis, B1, nNB);
  k_gcn_scatter<<<gE, blk, 0, stream>>>(B3, row, colv, ew, dis, B1, E);
  k_bias_relu<<<g1(nNB), blk, 0, stream>>>(B1, bg2, nNB);

  // --- hyper branch input: B2 = x * nhw ---
  k_scale_rows<<<g1(nNB), blk, 0, stream>>>(x, nhw, B2, nNB);
  // --- Hyper layer 1: B2 = relu(Dinv * H B^-1 H^T (B2 @ Wh1) + bh1) ---
  k_gemm_wmma<<<gGemm, blk, 0, stream>>>(B2, nullptr, 128, 0, Wh1t, nullptr, B3, 0, N);
  k_fill<<<g1(nMB), blk, 0, stream>>>(ef, 0.f, nMB);
  k_scatter_n2e<<<gI, blk, 0, stream>>>(B3, nodei, hei, ef, I);
  k_fill<<<g1(nNB), blk, 0, stream>>>(B2, 0.f, nNB);
  k_scatter_e2n<<<gI, blk, 0, stream>>>(ef, Binv, nodei, hei, B2, I);
  k_hyper_fin<<<g1(nNB), blk, 0, stream>>>(B2, Dinv, bh1, nNB);
  // --- Hyper layer 2 -> x2 in B2 ---
  k_gemm_wmma<<<gGemm, blk, 0, stream>>>(B2, nullptr, 128, 0, Wh2t, nullptr, B3, 0, N);
  k_fill<<<g1(nMB), blk, 0, stream>>>(ef, 0.f, nMB);
  k_scatter_n2e<<<gI, blk, 0, stream>>>(B3, nodei, hei, ef, I);
  k_fill<<<g1(nNB), blk, 0, stream>>>(B2, 0.f, nNB);
  k_scatter_e2n<<<gI, blk, 0, stream>>>(ef, Binv, nodei, hei, B2, I);
  k_hyper_fin<<<g1(nNB), blk, 0, stream>>>(B2, Dinv, bh2, nNB);

  // --- fused concat + FC: B3 = relu([x1|x2] @ Wfc + bfc), K = 256 ---
  k_gemm_wmma<<<gGemm, blk, 0, stream>>>(B1, B2, 128, 128, Wfct, bfc, B3, 1, N);

  // --- output projection 128 -> 2 ---
  k_outproj<<<g1(N), blk, 0, stream>>>(B3, Wo, bo, (float*)d_out, N);
}
